// DWTenhance_0_48765058679338
// MI455X (gfx1250) — matrix operand order repaired and checked by reading, not compiled, not run on hardware
//
#include <hip/hip_runtime.h>
#include <hip/hip_bf16.h>

typedef __attribute__((ext_vector_type(16))) __bf16 v16bf;
typedef __attribute__((ext_vector_type(8)))  float  v8f;

union BF16x16 { v16bf v; unsigned short s[16]; uint4 q[2]; };

__device__ __forceinline__ unsigned short f2bf(float f) {
  union { float f; unsigned int u; } cv; cv.f = f;
  unsigned int u = cv.u;
  u += 0x7FFFu + ((u >> 16) & 1u);   // round-to-nearest-even
  return (unsigned short)(u >> 16);
}

// ---------------------------------------------------------------------------
// Weight f32 -> bf16 pre-conversion (keeps cvt chains out of GEMM hot loops)
// ---------------------------------------------------------------------------
__global__ void w2bf_kernel(const float* __restrict__ src, unsigned short* __restrict__ dst,
                            unsigned int n) {
  unsigned int i = blockIdx.x * blockDim.x + threadIdx.x;
  if (i < n) dst[i] = f2bf(src[i]);
}

// ---------------------------------------------------------------------------
// DWT: img (B,64,256,256) f32 -> xT_ll [B][N][64] bf16, xT_hh [B][N][192] bf16
// ---------------------------------------------------------------------------
__global__ void dwt_kernel(const float* __restrict__ img,
                           unsigned short* __restrict__ xT_ll,
                           unsigned short* __restrict__ xT_hh) {
  int tid = blockIdx.x * blockDim.x + threadIdx.x;   // 2^24 threads
  int x = tid & 127;
  int y = (tid >> 7) & 127;
  int c = (tid >> 14) & 63;
  int b = tid >> 20;
  const float* p = img + (((size_t)(b * 64 + c) * 256 + 2 * y) * 256 + 2 * x);
  float a = p[0], bb = p[1], cc = p[256], d = p[257];
  float ll = (a + bb + cc + d) * 0.5f;
  float h1 = (a - bb + cc - d) * 0.5f;
  float h2 = (a + bb - cc - d) * 0.5f;
  float h3 = (a - bb - cc + d) * 0.5f;
  int n = (y << 7) | x;
  xT_ll[((size_t)b * 16384 + n) * 64 + c] = f2bf(ll);
  size_t hh = ((size_t)b * 16384 + n) * 192 + 3 * c;
  xT_hh[hh + 0] = f2bf(h1);
  xT_hh[hh + 1] = f2bf(h2);
  xT_hh[hh + 2] = f2bf(h3);
}

// ---------------------------------------------------------------------------
// Projection GEMM: dst = W[C,C] @ x[C,N] + bias ; one wave = 16x64 tile.
// Explicit double-buffered software pipeline; amdgpu_waves_per_eu relaxes the
// backend's occupancy/pressure target so the prefetch order is preserved.
// transposed==0 -> dst bf16 [c][N] (q,k) ; ==1 -> dst bf16 [n][C] (v).
// ---------------------------------------------------------------------------
template <int C>
__global__ __launch_bounds__(32)
__attribute__((amdgpu_waves_per_eu(1, 4)))
void proj_gemm(const unsigned short* __restrict__ Wbf, const float* __restrict__ bias,
               const unsigned short* __restrict__ xT,
               unsigned short* __restrict__ dst, int transposed,
               float* __restrict__ nsq, int HW) {
  constexpr int KS = C / 32;
  const int lane = threadIdx.x;
  const int ln = lane & 15;
  const int hi = lane >> 4;
  const int ntile = blockIdx.x * 64;
  const int mtile = blockIdx.y * 16;
  const int b = blockIdx.z;

  const unsigned short* xb = xT + (size_t)b * HW * C;
  const unsigned short* wr0 = Wbf + (size_t)(mtile + ln) * C + hi * 8;

  auto loadK = [&](int ks, BF16x16& A, BF16x16 (&Bm)[4]) {
    const unsigned short* ar = wr0 + ks * 32;
    A.q[0] = *(const uint4*)(ar);
    A.q[1] = *(const uint4*)(ar + 16);
    const int koff = ks * 32 + hi * 8;
#pragma unroll
    for (int s = 0; s < 4; ++s) {
      const uint4* bp = (const uint4*)(xb + (size_t)(ntile + s * 16 + ln) * C + koff);
      Bm[s].q[0] = bp[0];
      Bm[s].q[1] = bp[2];
    }
  };

  v8f cinit;
#pragma unroll
  for (int r = 0; r < 8; ++r) cinit[r] = bias[mtile + r + 8 * hi];
  v8f acc[4];
#pragma unroll
  for (int s = 0; s < 4; ++s) acc[s] = cinit;

  BF16x16 a[2];
  BF16x16 bm[2][4];
  loadK(0, a[0], bm[0]);
#pragma unroll
  for (int ks = 0; ks < KS; ++ks) {
    const int cur = ks & 1;
    if (ks + 1 < KS) loadK(ks + 1, a[cur ^ 1], bm[cur ^ 1]);
#pragma unroll
    for (int s = 0; s < 4; ++s)
      acc[s] = __builtin_amdgcn_wmma_f32_16x16x32_bf16(false, a[cur].v, false, bm[cur][s].v,
                                                       (short)0, acc[s], false, false);
  }

  if (!transposed) {
    unsigned short* db = dst + (size_t)b * C * HW;
#pragma unroll
    for (int s = 0; s < 4; ++s) {
      int n = ntile + s * 16 + ln;
#pragma unroll
      for (int r = 0; r < 8; ++r)
        db[(size_t)(mtile + r + 8 * hi) * HW + n] = f2bf(acc[s][r]);
    }
  } else {
    unsigned short* db = dst + (size_t)b * HW * C;
#pragma unroll
    for (int s = 0; s < 4; ++s) {
      int n = ntile + s * 16 + ln;
      uint4 pk;
      pk.x = (unsigned int)f2bf(acc[s][0]) | ((unsigned int)f2bf(acc[s][1]) << 16);
      pk.y = (unsigned int)f2bf(acc[s][2]) | ((unsigned int)f2bf(acc[s][3]) << 16);
      pk.z = (unsigned int)f2bf(acc[s][4]) | ((unsigned int)f2bf(acc[s][5]) << 16);
      pk.w = (unsigned int)f2bf(acc[s][6]) | ((unsigned int)f2bf(acc[s][7]) << 16);
      *(uint4*)(db + (size_t)n * C + mtile + 8 * hi) = pk;
    }
  }

  if (nsq) {
#pragma unroll
    for (int r = 0; r < 8; ++r) {
      float p = 0.f;
#pragma unroll
      for (int s = 0; s < 4; ++s) p += acc[s][r] * acc[s][r];
      p += __shfl_xor(p, 1);
      p += __shfl_xor(p, 2);
      p += __shfl_xor(p, 4);
      p += __shfl_xor(p, 8);
      if (ln == 0) atomicAdd(&nsq[(size_t)b * C + mtile + r + 8 * hi], p);
    }
  }
}

// ---------------------------------------------------------------------------
// Gram: S[C,C] += q . k^T over an N-chunk of KCH. Explicit two-phase rotation
// (64-K halves) with 4 independent accumulator chains.
// ---------------------------------------------------------------------------
template <int C, int KCH>
__global__ __launch_bounds__(32)
__attribute__((amdgpu_waves_per_eu(1, 4)))
void gram_gemm(const unsigned short* __restrict__ q,
               const unsigned short* __restrict__ k,
               float* __restrict__ S, int HW) {
  const int lane = threadIdx.x;
  const int ln = lane & 15;
  const int hi = lane >> 4;
  constexpr int T16 = C / 16;
  const int mtile = (blockIdx.y / T16) * 16;
  const int ntile = (blockIdx.y % T16) * 16;
  const int b = blockIdx.z;
  const int n0 = blockIdx.x * KCH;

  const unsigned short* qr = q + (size_t)b * C * HW + (size_t)(mtile + ln) * HW;
  const unsigned short* kr = k + (size_t)b * C * HW + (size_t)(ntile + ln) * HW;

  auto LD = [&](int base, BF16x16 (&A)[2], BF16x16 (&Bm)[2]) {
#pragma unroll
    for (int u = 0; u < 2; ++u) {
      int ab = base + u * 32 + hi * 8;
      A[u].q[0] = *(const uint4*)(qr + ab);
      A[u].q[1] = *(const uint4*)(qr + ab + 16);
      int bb = base + u * 32 + hi * 8;
      Bm[u].q[0] = *(const uint4*)(kr + bb);
      Bm[u].q[1] = *(const uint4*)(kr + bb + 16);
    }
  };

  constexpr int ITERS = KCH / 64;      // 64 K per half
  v8f acc0 = {}, acc1 = {}, acc2 = {}, acc3 = {};
  BF16x16 a0[2], b0[2], a1[2], b1[2];
  LD(n0, a0, b0);
  for (int i = 0; i < ITERS - 2; i += 2) {
    LD(n0 + (i + 1) * 64, a1, b1);
    acc0 = __builtin_amdgcn_wmma_f32_16x16x32_bf16(false, a0[0].v, false, b0[0].v, (short)0, acc0, false, false);
    acc1 = __builtin_amdgcn_wmma_f32_16x16x32_bf16(false, a0[1].v, false, b0[1].v, (short)0, acc1, false, false);
    LD(n0 + (i + 2) * 64, a0, b0);
    acc2 = __builtin_amdgcn_wmma_f32_16x16x32_bf16(false, a1[0].v, false, b1[0].v, (short)0, acc2, false, false);
    acc3 = __builtin_amdgcn_wmma_f32_16x16x32_bf16(false, a1[1].v, false, b1[1].v, (short)0, acc3, false, false);
  }
  LD(n0 + (ITERS - 1) * 64, a1, b1);
  acc0 = __builtin_amdgcn_wmma_f32_16x16x32_bf16(false, a0[0].v, false, b0[0].v, (short)0, acc0, false, false);
  acc1 = __builtin_amdgcn_wmma_f32_16x16x32_bf16(false, a0[1].v, false, b0[1].v, (short)0, acc1, false, false);
  acc2 = __builtin_amdgcn_wmma_f32_16x16x32_bf16(false, a1[0].v, false, b1[0].v, (short)0, acc2, false, false);
  acc3 = __builtin_amdgcn_wmma_f32_16x16x32_bf16(false, a1[1].v, false, b1[1].v, (short)0, acc3, false, false);

  v8f acc = (acc0 + acc1) + (acc2 + acc3);
  float* Sb = S + (size_t)b * C * C;
#pragma unroll
  for (int r = 0; r < 8; ++r)
    atomicAdd(&Sb[(size_t)(mtile + r + 8 * hi) * C + ntile + ln], acc[r]);
}

// ---------------------------------------------------------------------------
// Per-batch: scale by 1/(|q||k|)*temp, softmax rows in-place, then fold the
// output projection: M2 = pw @ attn (bf16). One block (C threads) per batch.
// ---------------------------------------------------------------------------
__global__ void softmax_m2(float* __restrict__ S, const float* __restrict__ nq,
                           const float* __restrict__ nk, const float* __restrict__ temp,
                           const float* __restrict__ pw, unsigned short* __restrict__ M2,
                           int C) {
  const int b = blockIdx.x;
  const int t = threadIdx.x;
  float* Sb = S + (size_t)b * C * C;
  const float* nqb = nq + (size_t)b * C;
  const float* nkb = nk + (size_t)b * C;
  const float tmp = temp[0];
  const float EPS = 1e-12f;

  float inq = 1.0f / fmaxf(sqrtf(fmaxf(nqb[t], 0.f)), EPS);
  float mx = -3.4e38f;
  for (int d = 0; d < C; ++d) {
    float ink = 1.0f / fmaxf(sqrtf(fmaxf(nkb[d], 0.f)), EPS);
    float l = Sb[(size_t)t * C + d] * inq * ink * tmp;
    Sb[(size_t)t * C + d] = l;
    mx = fmaxf(mx, l);
  }
  float sum = 0.f;
  for (int d = 0; d < C; ++d) {
    float e = __expf(Sb[(size_t)t * C + d] - mx);
    Sb[(size_t)t * C + d] = e;
    sum += e;
  }
  float inv = 1.0f / sum;
  for (int d = 0; d < C; ++d) Sb[(size_t)t * C + d] *= inv;

  __threadfence();
  __syncthreads();

  unsigned short* M2b = M2 + (size_t)b * C * C;
  for (int d = 0; d < C; ++d) {
    float s = 0.f;
    for (int c = 0; c < C; ++c) s += pw[(size_t)t * C + c] * Sb[(size_t)c * C + d];
    M2b[(size_t)t * C + d] = f2bf(s);
  }
}

// ---------------------------------------------------------------------------
// Output GEMM: Y[C,N] = M2[C,C] @ v[C,N] + pb. Same double-buffered pipeline.
// ---------------------------------------------------------------------------
template <int C>
__global__ __launch_bounds__(32)
__attribute__((amdgpu_waves_per_eu(1, 4)))
void out_gemm(const unsigned short* __restrict__ M2, const float* __restrict__ pb,
              const unsigned short* __restrict__ v, float* __restrict__ Y,
              int HW) {
  constexpr int KS = C / 32;
  const int lane = threadIdx.x;
  const int ln = lane & 15;
  const int hi = lane >> 4;
  const int ntile = blockIdx.x * 64;
  const int mtile = blockIdx.y * 16;
  const int b = blockIdx.z;

  const unsigned short* M2r = M2 + (size_t)b * C * C + (size_t)(mtile + ln) * C + hi * 8;
  const unsigned short* vb  = v + (size_t)b * HW * C;

  auto loadK = [&](int ks, BF16x16& A, BF16x16 (&Bm)[4]) {
    A.q[0] = *(const uint4*)(M2r + ks * 32);
    A.q[1] = *(const uint4*)(M2r + ks * 32 + 16);
    const int koff = ks * 32 + hi * 8;
#pragma unroll
    for (int s = 0; s < 4; ++s) {
      const uint4* bp = (const uint4*)(vb + (size_t)(ntile + s * 16 + ln) * C + koff);
      Bm[s].q[0] = bp[0];
      Bm[s].q[1] = bp[2];
    }
  };

  v8f cinit;
#pragma unroll
  for (int r = 0; r < 8; ++r) cinit[r] = pb[mtile + r + 8 * hi];
  v8f acc[4];
#pragma unroll
  for (int s = 0; s < 4; ++s) acc[s] = cinit;

  BF16x16 a[2];
  BF16x16 bm[2][4];
  loadK(0, a[0], bm[0]);
#pragma unroll
  for (int ks = 0; ks < KS; ++ks) {
    const int cur = ks & 1;
    if (ks + 1 < KS) loadK(ks + 1, a[cur ^ 1], bm[cur ^ 1]);
#pragma unroll
    for (int s = 0; s < 4; ++s)
      acc[s] = __builtin_amdgcn_wmma_f32_16x16x32_bf16(false, a[cur].v, false, bm[cur][s].v,
                                                       (short)0, acc[s], false, false);
  }

  float* Yb = Y + (size_t)b * C * HW;
#pragma unroll
  for (int s = 0; s < 4; ++s) {
    int n = ntile + s * 16 + ln;
#pragma unroll
    for (int r = 0; r < 8; ++r)
      Yb[(size_t)(mtile + r + 8 * hi) * HW + n] = acc[s][r];
  }
}

// ---------------------------------------------------------------------------
// iDWT: Yll (B,64,N) + Yhh (B,192,N) f32 -> out (B,64,256,256) f32
// ---------------------------------------------------------------------------
__global__ void idwt_kernel(const float* __restrict__ Yll, const float* __restrict__ Yhh,
                            float* __restrict__ out) {
  int tid = blockIdx.x * blockDim.x + threadIdx.x;
  int x = tid & 127;
  int y = (tid >> 7) & 127;
  int c = (tid >> 14) & 63;
  int b = tid >> 20;
  int n = (y << 7) | x;
  float ll = Yll[(size_t)(b * 64 + c) * 16384 + n];
  size_t hb = (size_t)(b * 192 + 3 * c) * 16384 + n;
  float h1 = Yhh[hb];
  float h2 = Yhh[hb + 16384];
  float h3 = Yhh[hb + 2 * 16384];
  float a  = (ll + h1 + h2 + h3) * 0.5f;
  float bb = (ll - h1 + h2 - h3) * 0.5f;
  float cc = (ll + h1 - h2 - h3) * 0.5f;
  float d  = (ll - h1 - h2 + h3) * 0.5f;
  float* p = out + (((size_t)(b * 64 + c) * 256 + 2 * y) * 256 + 2 * x);
  p[0] = a; p[1] = bb; p[256] = cc; p[257] = d;
}

__global__ void zero_kernel(float* __restrict__ p, unsigned int n) {
  unsigned int i = blockIdx.x * blockDim.x + threadIdx.x;
  if (i < n) p[i] = 0.f;
}

// ---------------------------------------------------------------------------
extern "C" void kernel_launch(void* const* d_in, const int* in_sizes, int n_in,
                              void* d_out, int out_size, void* d_ws, size_t ws_size,
                              hipStream_t stream) {
  const float* img   = (const float*)d_in[0];
  const float* ll_qw = (const float*)d_in[1];
  const float* h_qw  = (const float*)d_in[2];
  const float* ll_qb = (const float*)d_in[3];
  const float* h_qb  = (const float*)d_in[4];
  const float* ll_kw = (const float*)d_in[5];
  const float* h_kw  = (const float*)d_in[6];
  const float* ll_kb = (const float*)d_in[7];
  const float* h_kb  = (const float*)d_in[8];
  const float* ll_vw = (const float*)d_in[9];
  const float* h_vw  = (const float*)d_in[10];
  const float* ll_vb = (const float*)d_in[11];
  const float* h_vb  = (const float*)d_in[12];
  const float* ll_pw = (const float*)d_in[13];
  const float* h_pw  = (const float*)d_in[14];
  const float* ll_pb = (const float*)d_in[15];
  const float* h_pb  = (const float*)d_in[16];
  const float* ll_t  = (const float*)d_in[17];
  const float* h_t   = (const float*)d_in[18];

  const int B = 16, HW = 16384;
  char* ws = (char*)d_ws;
  size_t off = 0;
  auto carve = [&](size_t bytes) -> void* {
    void* p = ws + off;
    off = (off + bytes + 255) & ~(size_t)255;
    return p;
  };

  unsigned short* xT_ll = (unsigned short*)carve((size_t)B * HW * 64 * 2);
  unsigned short* xT_hh = (unsigned short*)carve((size_t)B * HW * 192 * 2);
  unsigned short* qbuf  = (unsigned short*)carve((size_t)B * 192 * HW * 2);
  unsigned short* kbuf  = (unsigned short*)carve((size_t)B * 192 * HW * 2);
  unsigned short* vbuf  = (unsigned short*)carve((size_t)B * HW * 192 * 2);
  float* Sbuf = (float*)carve((size_t)B * 192 * 192 * 4);
  float* nqb  = (float*)carve((size_t)B * 192 * 4);
  float* nkb  = (float*)carve((size_t)B * 192 * 4);
  unsigned short* M2b = (unsigned short*)carve((size_t)B * 192 * 192 * 2);
  unsigned short* wqb = (unsigned short*)carve((size_t)192 * 192 * 2);
  unsigned short* wkb = (unsigned short*)carve((size_t)192 * 192 * 2);
  unsigned short* wvb = (unsigned short*)carve((size_t)192 * 192 * 2);
  float* Yll = (float*)carve((size_t)B * 64 * HW * 4);
  float* Yhh = (float*)carve((size_t)B * 192 * HW * 4);

  dwt_kernel<<<(B * 64 * HW) / 256, 256, 0, stream>>>(img, xT_ll, xT_hh);

  struct Stage {
    int C;
    const unsigned short* xT;
    const float *qw, *qb, *kw, *kb, *vw, *vb, *pw, *pb, *temp;
    float* Y;
  } stages[2] = {
    { 64, xT_ll, ll_qw, ll_qb, ll_kw, ll_kb, ll_vw, ll_vb, ll_pw, ll_pb, ll_t, Yll},
    {192, xT_hh, h_qw,  h_qb,  h_kw,  h_kb,  h_vw,  h_vb,  h_pw,  h_pb,  h_t,  Yhh},
  };

  for (int si = 0; si < 2; ++si) {
    const Stage& st = stages[si];
    const int C = st.C;
    const unsigned int cc = (unsigned int)(C * C);

    // bf16 weight copies
    w2bf_kernel<<<(cc + 255) / 256, 256, 0, stream>>>(st.qw, wqb, cc);
    w2bf_kernel<<<(cc + 255) / 256, 256, 0, stream>>>(st.kw, wkb, cc);
    w2bf_kernel<<<(cc + 255) / 256, 256, 0, stream>>>(st.vw, wvb, cc);

    unsigned int zs = (unsigned int)(B * 192 * 192);
    zero_kernel<<<(zs + 255) / 256, 256, 0, stream>>>(Sbuf, zs);
    zero_kernel<<<(B * 192 + 255) / 256, 256, 0, stream>>>(nqb, B * 192);
    zero_kernel<<<(B * 192 + 255) / 256, 256, 0, stream>>>(nkb, B * 192);

    dim3 gp(HW / 64, C / 16, B);
    constexpr int KCH = 2048;
    dim3 gg(HW / KCH, (C / 16) * (C / 16), B);
    dim3 go(HW / 64, C / 16, B);

    if (C == 64) {
      proj_gemm<64><<<gp, 32, 0, stream>>>(wqb, st.qb, st.xT, qbuf, 0, nqb, HW);
      proj_gemm<64><<<gp, 32, 0, stream>>>(wkb, st.kb, st.xT, kbuf, 0, nkb, HW);
      proj_gemm<64><<<gp, 32, 0, stream>>>(wvb, st.vb, st.xT, vbuf, 1, nullptr, HW);
      gram_gemm<64, KCH><<<gg, 32, 0, stream>>>(qbuf, kbuf, Sbuf, HW);
      softmax_m2<<<B, C, 0, stream>>>(Sbuf, nqb, nkb, st.temp, st.pw, M2b, C);
      out_gemm<64><<<go, 32, 0, stream>>>(M2b, st.pb, vbuf, st.Y, HW);
    } else {
      proj_gemm<192><<<gp, 32, 0, stream>>>(wqb, st.qb, st.xT, qbuf, 0, nqb, HW);
      proj_gemm<192><<<gp, 32, 0, stream>>>(wkb, st.kb, st.xT, kbuf, 0, nkb, HW);
      proj_gemm<192><<<gp, 32, 0, stream>>>(wvb, st.vb, st.xT, vbuf, 1, nullptr, HW);
      gram_gemm<192, KCH><<<gg, 32, 0, stream>>>(qbuf, kbuf, Sbuf, HW);
      softmax_m2<<<B, C, 0, stream>>>(Sbuf, nqb, nkb, st.temp, st.pw, M2b, C);
      out_gemm<192><<<go, 32, 0, stream>>>(M2b, st.pb, vbuf, st.Y, HW);
    }
  }

  idwt_kernel<<<(B * 64 * HW) / 256, 256, 0, stream>>>(Yll, Yhh, (float*)d_out);
}
